// GraphAttention_833223656413
// MI455X (gfx1250) — matrix operand order repaired and checked by reading, not compiled, not run on hardware
//
#include <hip/hip_runtime.h>
#include <hip/hip_bf16.h>
#include <math.h>

// MI455X / gfx1250 fused equivariant graph attention.
// B=1, N=2048, K=32, D=128, H=8, DH=16, fp32 in/out.
//
// - Dominant GEMM (edge_emb @ Wdtp, 8.6 GFLOP) on v_wmma_f32_16x16x32_f16
//   (f16 operands / f32 accumulate); short-K contractions stay on VALU which
//   co-executes with WMMA on CDNA5.
// - edge_emb tile staged into LDS with GLOBAL_LOAD_ASYNC_TO_LDS_B128 (inline
//   asm, ASYNCcnt) overlapping the qrow load + kproj prefetch.
// - One workgroup (8 wave32 = one wave per head) per query row; w staged in
//   LDS; softmax + K-reduction via wave32 shuffles (lane = neighbor).

typedef _Float16 v16h __attribute__((ext_vector_type(16)));
typedef float    v8f  __attribute__((ext_vector_type(8)));

#define NQ   2048
#define KN   32
#define DD   128
#define HH   8
#define DHH  16
#define EPSF 1e-5f
#define FMIN32 (-3.402823466e38f)

__device__ __forceinline__ float wave_sum32(float v) {
#pragma unroll
  for (int o = 16; o > 0; o >>= 1) v += __shfl_xor(v, o, 32);
  return v;
}
__device__ __forceinline__ float wave_max32(float v) {
#pragma unroll
  for (int o = 16; o > 0; o >>= 1) v = fmaxf(v, __shfl_xor(v, o, 32));
  return v;
}

// Async DMA: 16B global -> LDS, tracked by ASYNCcnt (gfx1250).
// VGLOBAL encoding: VDST = LDS byte offset VGPR, VADDR = 64-bit global addr.
__device__ __forceinline__ void async_g2lds_b128(unsigned int lds_off, const void* g) {
  asm volatile("global_load_async_to_lds_b128 %0, %1, off"
               :: "v"(lds_off), "v"((unsigned long long)(size_t)g)
               : "memory");
}
__device__ __forceinline__ void wait_async0() {
  asm volatile("s_wait_asynccnt 0" ::: "memory");
}

// ---------------------------------------------------------------- prep: Wdtp -> f16
__global__ void cvt_wdtp(const float* __restrict__ w, _Float16* __restrict__ o) {
  int i = blockIdx.x * 256 + threadIdx.x;
  if (i < DD * 4 * DD) o[i] = (_Float16)w[i];
}

// ---------------------------------------------------------------- LN + q/k projections
__global__ void ln_proj(const float* __restrict__ qs_in, const float* __restrict__ qv_in,
                        const float* __restrict__ ks_in, const float* __restrict__ kv_in,
                        const float* __restrict__ Wq_s, const float* __restrict__ Wq_v,
                        const float* __restrict__ bq,
                        const float* __restrict__ Wk_s, const float* __restrict__ Wk_v,
                        const float* __restrict__ bk,
                        const float* __restrict__ ln_gs, const float* __restrict__ ln_bs,
                        const float* __restrict__ ln_gv,
                        float* __restrict__ ln_s, float* __restrict__ ln_v,
                        float* __restrict__ qproj, float* __restrict__ kproj) {
  __shared__ float sh_s[DD];
  __shared__ float sh_v[DD * 3];
  __shared__ float red[DD];
  const int n = blockIdx.x, t = threadIdx.x;

  // ---- equivariant layer norm on the query ----
  float s = qs_in[n * DD + t];
  red[t] = s; __syncthreads();
  for (int o = 64; o > 0; o >>= 1) { if (t < o) red[t] += red[t + o]; __syncthreads(); }
  float mu = red[0] * (1.f / DD); __syncthreads();
  float dv = s - mu;
  red[t] = dv * dv; __syncthreads();
  for (int o = 64; o > 0; o >>= 1) { if (t < o) red[t] += red[t + o]; __syncthreads(); }
  float var = red[0] * (1.f / DD); __syncthreads();
  float sl = dv * rsqrtf(var + EPSF) * ln_gs[t] + ln_bs[t];

  float v0 = qv_in[(n * DD + t) * 3 + 0];
  float v1 = qv_in[(n * DD + t) * 3 + 1];
  float v2 = qv_in[(n * DD + t) * 3 + 2];
  red[t] = v0 * v0 + v1 * v1 + v2 * v2; __syncthreads();
  for (int o = 64; o > 0; o >>= 1) { if (t < o) red[t] += red[t + o]; __syncthreads(); }
  float rms = sqrtf(red[0] * (1.f / DD) + EPSF); __syncthreads();
  float gv = ln_gv[t] / rms;
  float w0 = v0 * gv, w1 = v1 * gv, w2 = v2 * gv;

  sh_s[t] = sl;
  sh_v[t * 3 + 0] = w0; sh_v[t * 3 + 1] = w1; sh_v[t * 3 + 2] = w2;
  ln_s[n * DD + t] = sl;                       // skip connection (post-LN)
  ln_v[(n * DD + t) * 3 + 0] = w0;
  ln_v[(n * DD + t) * 3 + 1] = w1;
  ln_v[(n * DD + t) * 3 + 2] = w2;
  __syncthreads();

  // ---- q projection: LN(query) @ Wq ----
  {
    float accs = bq[t], a0 = 0.f, a1 = 0.f, a2 = 0.f;
    for (int i = 0; i < DD; i++) {
      float w_ = Wq_s[i * DD + t], wv_ = Wq_v[i * DD + t];
      accs += sh_s[i] * w_;
      a0 += sh_v[i * 3 + 0] * wv_;
      a1 += sh_v[i * 3 + 1] * wv_;
      a2 += sh_v[i * 3 + 2] * wv_;
    }
    qproj[n * 512 + t] = accs;
    qproj[n * 512 + DD + t * 3 + 0] = a0;
    qproj[n * 512 + DD + t * 3 + 1] = a1;
    qproj[n * 512 + DD + t * 3 + 2] = a2;
  }
  __syncthreads();

  // ---- k projection: raw key @ Wk ----
  sh_s[t] = ks_in[n * DD + t];
  sh_v[t * 3 + 0] = kv_in[(n * DD + t) * 3 + 0];
  sh_v[t * 3 + 1] = kv_in[(n * DD + t) * 3 + 1];
  sh_v[t * 3 + 2] = kv_in[(n * DD + t) * 3 + 2];
  __syncthreads();
  {
    float accs = bk[t], a0 = 0.f, a1 = 0.f, a2 = 0.f;
    for (int i = 0; i < DD; i++) {
      float w_ = Wk_s[i * DD + t], wv_ = Wk_v[i * DD + t];
      accs += sh_s[i] * w_;
      a0 += sh_v[i * 3 + 0] * wv_;
      a1 += sh_v[i * 3 + 1] * wv_;
      a2 += sh_v[i * 3 + 2] * wv_;
    }
    kproj[n * 512 + t] = accs;
    kproj[n * 512 + DD + t * 3 + 0] = a0;
    kproj[n * 512 + DD + t * 3 + 1] = a1;
    kproj[n * 512 + DD + t * 3 + 2] = a2;
  }
}

// ---------------------------------------------------------------- fused attention
__global__ void __launch_bounds__(256)
fused_attn(const float* __restrict__ qproj, const float* __restrict__ kproj,
           const float* __restrict__ ln_s, const float* __restrict__ ln_v,
           const float* __restrict__ eir_s, const float* __restrict__ eir_v,
           const float* __restrict__ edge_emb,
           const int* __restrict__ mask, const int* __restrict__ knn,
           const _Float16* __restrict__ wdtp_h, const float* __restrict__ bdtp,
           const float* __restrict__ Wqk_s, const float* __restrict__ Wqk_v,
           const float* __restrict__ bqk,
           const float* __restrict__ Wagv_s, const float* __restrict__ bagv,
           const float* __restrict__ Wagv_v,
           const float* __restrict__ Wattn,
           const float* __restrict__ lna_g, const float* __restrict__ lna_b,
           const float* __restrict__ Wval_s, const float* __restrict__ Wval_v,
           const float* __restrict__ bval,
           const float* __restrict__ Wo_s, const float* __restrict__ Wo_v,
           const float* __restrict__ bo,
           float* __restrict__ out) {
  // dynamic LDS carve:
  //  E32 [32*128] f32 : 16384 B   (async-DMA staging)
  //  E16 [32*128] f16 :  8192 B   (WMMA A operand)
  //  w   [32*512] f32 : 65536 B
  //  qrow[512]    f32 :  2048 B
  //  outpre[512]  f32 :  2048 B     total 94208 B
  extern __shared__ unsigned char smem_raw[];
  float*    E32    = (float*)smem_raw;
  _Float16* E16    = (_Float16*)(smem_raw + 16384);
  float*    w_lds  = (float*)(smem_raw + 16384 + 8192);
  float*    qrow   = w_lds + KN * 512;
  float*    outpre = qrow + 512;

  const int n = blockIdx.x, t = threadIdx.x;
  const int wave = t >> 5, lane = t & 31;
  const int h = wave, k = lane;
  const int nk = n * KN + k;

  // ---- phase 0: async-DMA edge_emb tile into LDS; overlap qrow + prefetch ----
  {
    const float* src = edge_emb + (size_t)n * KN * DD;     // 16 KB tile
    unsigned int lbase = (unsigned int)(size_t)E32;        // LDS byte offset
#pragma unroll
    for (int c = 0; c < 4; c++) {
      int i = t + c * 256;                                  // 1024 x 16B chunks
      async_g2lds_b128(lbase + i * 16, (const char*)src + i * 16);
    }
  }
  const int nbr = knn[nk];
  const float* kp = kproj + (size_t)nbr * 512;              // gathered key row
  {
    // warm the L2->WGP path for the slices phase 2 will read
    int base = (h < 4 ? h : h - 4) * 32;
    __builtin_prefetch(kp + base, 0, 1);
    __builtin_prefetch(kp + DD + base * 3, 0, 1);
  }
  for (int i = t; i < 512; i += 256) qrow[i] = qproj[n * 512 + i];
  wait_async0();
  __syncthreads();
  // convert staged tile to f16 once (WMMA A operand)
  for (int i = t; i < KN * DD; i += 256) E16[i] = (_Float16)E32[i];
  __syncthreads();

  // ---- phase 1: w = E(32x128) @ Wdtp(128x512) + bdtp via WMMA f16 -> f32 ----
  // 64 tiles (2 M x 32 N), 8 per wave; uniform SGPR loop keeps EXEC all-ones.
  const int wave_u = __builtin_amdgcn_readfirstlane(wave);
  for (int tile = wave_u; tile < 64; tile += 8) {
    int ntile = tile & 31, mtile = tile >> 5;
    v8f c = {0.f, 0.f, 0.f, 0.f, 0.f, 0.f, 0.f, 0.f};
#pragma unroll
    for (int kst = 0; kst < 4; kst++) {
      v16h a, b;
      // A fragment (16x32, ISA layout: lane = M (+bank), halves: K 0..7 / 16..23 (+8))
      int m  = mtile * 16 + (lane & 15);
      int kb = kst * 32 + ((lane >> 4) << 3);
#pragma unroll
      for (int j = 0; j < 16; j++)
        a[j] = E16[m * DD + kb + (j & 7) + ((j >> 3) << 4)];
      // B fragment (32x16, ISA layout: lane = N (+bank), halves: K 0..7 / 16..23 (+8))
      const _Float16* bp = wdtp_h + ntile * 16 + (lane & 15);
#pragma unroll
      for (int j = 0; j < 16; j++) b[j] = bp[(kb + (j & 7) + ((j >> 3) << 4)) * 512];
      c = __builtin_amdgcn_wmma_f32_16x16x32_f16(false, a, false, b, (short)0, c,
                                                 false, false);
    }
    // C/D layout: VGPR r, lane l -> M = r + 8*(l>=16), N = l%16
    int rb  = mtile * 16 + ((lane >> 4) << 3);
    int col = ntile * 16 + (lane & 15);
    float bc = bdtp[col];
#pragma unroll
    for (int r = 0; r < 8; r++) w_lds[(rb + r) * 512 + col] = c[r] + bc;
  }
  __syncthreads();

  // ---- phase 2: per-(head, neighbor) pipeline; wave = head, lane = neighbor ----
  const float es  = eir_s[nk];
  const float ev0 = eir_v[nk * 3 + 0];
  const float ev1 = eir_v[nk * 3 + 1];
  const float ev2 = eir_v[nk * 3 + 2];

  // dtp1 (q x gathered-k) streamed directly into the Wqk 32->16 contraction.
  float qks[DHH];
  float qkv[DHH * 3];
#pragma unroll
  for (int d = 0; d < DHH; d++) {
    qks[d] = bqk[d];
    qkv[d * 3 + 0] = 0.f; qkv[d * 3 + 1] = 0.f; qkv[d * 3 + 2] = 0.f;
  }
  if (h < 4) {  // heads 0..3: scalar part = ss, vector part = sv
    const int base = h * 32;
    for (int i = 0; i < 32; i++) {
      int j = base + i;
      float qsj = qrow[j];
      float ps  = qsj * kp[j];
      float pv0 = qsj * kp[DD + j * 3 + 0];
      float pv1 = qsj * kp[DD + j * 3 + 1];
      float pv2 = qsj * kp[DD + j * 3 + 2];
#pragma unroll
      for (int d = 0; d < DHH; d++) {
        float ws_ = Wqk_s[i * DHH + d], wv_ = Wqk_v[i * DHH + d];
        qks[d] += ps * ws_;
        qkv[d * 3 + 0] += pv0 * wv_;
        qkv[d * 3 + 1] += pv1 * wv_;
        qkv[d * 3 + 2] += pv2 * wv_;
      }
    }
  } else {      // heads 4..7: scalar part = vv, vector part = vs
    const int base = (h - 4) * 32;
    for (int i = 0; i < 32; i++) {
      int j = base + i;
      float q0 = qrow[DD + j * 3 + 0], q1 = qrow[DD + j * 3 + 1], q2 = qrow[DD + j * 3 + 2];
      float k0 = kp[DD + j * 3 + 0],   k1 = kp[DD + j * 3 + 1],   k2 = kp[DD + j * 3 + 2];
      float ps  = q0 * k0 + q1 * k1 + q2 * k2;
      float ksj = kp[j];
      float pv0 = ksj * q0, pv1 = ksj * q1, pv2 = ksj * q2;
#pragma unroll
      for (int d = 0; d < DHH; d++) {
        float ws_ = Wqk_s[i * DHH + d], wv_ = Wqk_v[i * DHH + d];
        qks[d] += ps * ws_;
        qkv[d * 3 + 0] += pv0 * wv_;
        qkv[d * 3 + 1] += pv1 * wv_;
        qkv[d * 3 + 2] += pv2 * wv_;
      }
    }
  }

  // dtp2 (weighted, with edge irreps) streamed into Wagv contractions.
  float agv[48];
  float vlv[48];
#pragma unroll
  for (int o = 0; o < 48; o++) { agv[o] = bagv[h * 48 + o]; vlv[o] = 0.f; }
  const float* Wgs = Wagv_s + h * 32 * 48;
  const float* Wgv = Wagv_v + h * 32 * 16;
  const float* wk  = w_lds + k * 512 + h * DHH;   // w[(n,k)][wi*128 + h*16 + d]
#pragma unroll
  for (int d = 0; d < DHH; d++) {
    float w0 = wk[d], w1 = wk[128 + d], w2 = wk[256 + d], w3 = wk[384 + d];
    float ss2 = qks[d] * es * w0;
    float vv2 = (qkv[d * 3] * ev0 + qkv[d * 3 + 1] * ev1 + qkv[d * 3 + 2] * ev2) * w1;
    const float* wa = Wgs + d * 48;
    const float* wb = Wgs + (16 + d) * 48;
#pragma unroll
    for (int o = 0; o < 48; o++) agv[o] += ss2 * wa[o] + vv2 * wb[o];
    float sva = qks[d] * w2;
    float sv0 = sva * ev0, sv1 = sva * ev1, sv2 = sva * ev2;
    float vsa = es * w3;
    float vs0 = vsa * qkv[d * 3], vs1 = vsa * qkv[d * 3 + 1], vs2 = vsa * qkv[d * 3 + 2];
    const float* ua = Wgv + d * 16;
    const float* ub = Wgv + (16 + d) * 16;
#pragma unroll
    for (int o = 0; o < 16; o++) {
      float u1 = ua[o], u2 = ub[o];
      vlv[o * 3 + 0] += sv0 * u1 + vs0 * u2;
      vlv[o * 3 + 1] += sv1 * u1 + vs1 * u2;
      vlv[o * 3 + 2] += sv2 * u1 + vs2 * u2;
    }
  }

  // attention logit: LN over DH then dot with Wattn; in-wave softmax over k.
  float mu = 0.f;
#pragma unroll
  for (int d = 0; d < DHH; d++) mu += agv[d];
  mu *= (1.f / DHH);
  float var = 0.f;
#pragma unroll
  for (int d = 0; d < DHH; d++) { float df = agv[d] - mu; var += df * df; }
  var *= (1.f / DHH);
  float inv = rsqrtf(var + EPSF);
  float logit = 0.f;
#pragma unroll
  for (int d = 0; d < DHH; d++)
    logit += ((agv[d] - mu) * inv * lna_g[d] + lna_b[d]) * Wattn[d];
  if (mask[nk] == 0) logit = FMIN32;
  float mx    = wave_max32(logit);
  float ee    = __expf(logit - mx);
  float ssum  = wave_sum32(ee);
  float attnw = ee / ssum;
  out[NQ * DD + NQ * DD * 3 + h * NQ * KN + nk] = attnw;   // attn output (H,B,N,K)

  // silu(val_s), sigmoid-gated val_v.
  float vals[DHH];
#pragma unroll
  for (int o = 0; o < DHH; o++) {
    float g  = agv[16 + o];
    float sg = 1.f / (1.f + __expf(-g));
    float v_ = agv[32 + o];
    vals[o] = v_ / (1.f + __expf(-v_));
    vlv[o * 3 + 0] *= sg; vlv[o * 3 + 1] *= sg; vlv[o * 3 + 2] *= sg;
  }

  // dtp3 (no weights) streamed into Wval 32->16; then attn-weight + wave K-sum.
  float fs[DHH];
  float fv[48];
#pragma unroll
  for (int o = 0; o < DHH; o++) fs[o] = bval[o];
#pragma unroll
  for (int o = 0; o < 48; o++) fv[o] = 0.f;
#pragma unroll
  for (int d = 0; d < DHH; d++) {
    float ss3 = vals[d] * es;
    float vv3 = vlv[d * 3] * ev0 + vlv[d * 3 + 1] * ev1 + vlv[d * 3 + 2] * ev2;
    float sv0 = vals[d] * ev0, sv1 = vals[d] * ev1, sv2 = vals[d] * ev2;
    float vs0 = es * vlv[d * 3], vs1 = es * vlv[d * 3 + 1], vs2 = es * vlv[d * 3 + 2];
    const float* w1p = Wval_s + d * 16;
    const float* w2p = Wval_s + (16 + d) * 16;
    const float* u1p = Wval_v + d * 16;
    const float* u2p = Wval_v + (16 + d) * 16;
#pragma unroll
    for (int o = 0; o < DHH; o++) {
      fs[o] += ss3 * w1p[o] + vv3 * w2p[o];
      float u1 = u1p[o], u2 = u2p[o];
      fv[o * 3 + 0] += sv0 * u1 + vs0 * u2;
      fv[o * 3 + 1] += sv1 * u1 + vs1 * u2;
      fv[o * 3 + 2] += sv2 * u1 + vs2 * u2;
    }
  }
#pragma unroll
  for (int o = 0; o < DHH; o++) {
    float r = wave_sum32(fs[o] * attnw);
    if (lane == 0) outpre[h * 16 + o] = r;
  }
#pragma unroll
  for (int oc = 0; oc < 48; oc++) {
    float r = wave_sum32(fv[oc] * attnw);
    if (lane == 0) outpre[DD + h * 48 + oc] = r;
  }
  __syncthreads();

  // ---- phase 3: final Wo projection + residual ----
  if (t < DD) {
    float accs = bo[t], a0 = 0.f, a1 = 0.f, a2 = 0.f;
    for (int i = 0; i < DD; i++) {
      float si = outpre[i];
      float w_ = Wo_s[i * DD + t];
      accs += si * w_;
      float wv = Wo_v[i * DD + t];
      a0 += outpre[DD + i * 3 + 0] * wv;
      a1 += outpre[DD + i * 3 + 1] * wv;
      a2 += outpre[DD + i * 3 + 2] * wv;
    }
    out[n * DD + t] = accs + ln_s[n * DD + t];
    int vb = NQ * DD + (n * DD + t) * 3;
    out[vb + 0] = a0 + ln_v[(n * DD + t) * 3 + 0];
    out[vb + 1] = a1 + ln_v[(n * DD + t) * 3 + 1];
    out[vb + 2] = a2 + ln_v[(n * DD + t) * 3 + 2];
  }
}

// ---------------------------------------------------------------- host
extern "C" void kernel_launch(void* const* d_in, const int* in_sizes, int n_in,
                              void* d_out, int out_size, void* d_ws, size_t ws_size,
                              hipStream_t stream) {
  (void)in_sizes; (void)n_in; (void)out_size; (void)ws_size;
  const float* query_s = (const float*)d_in[0];
  const float* query_v = (const float*)d_in[1];
  const float* key_s   = (const float*)d_in[2];
  const float* key_v   = (const float*)d_in[3];
  const float* eir_s   = (const float*)d_in[4];
  const float* eir_v   = (const float*)d_in[5];
  const float* edge_emb= (const float*)d_in[6];
  const int*   maskp   = (const int*)d_in[7];
  const int*   knn     = (const int*)d_in[8];
  const float* Wq_s = (const float*)d_in[9];
  const float* Wq_v = (const float*)d_in[10];
  const float* bq   = (const float*)d_in[11];
  const float* Wk_s = (const float*)d_in[12];
  const float* Wk_v = (const float*)d_in[13];
  const float* bk   = (const float*)d_in[14];
  const float* Wqk_s= (const float*)d_in[15];
  const float* Wqk_v= (const float*)d_in[16];
  const float* bqk  = (const float*)d_in[17];
  const float* Wdtp = (const float*)d_in[18];
  const float* bdtp = (const float*)d_in[19];
  const float* Wagv_s=(const float*)d_in[20];
  const float* bagv = (const float*)d_in[21];
  const float* Wagv_v=(const float*)d_in[22];
  const float* Wattn= (const float*)d_in[23];
  const float* Wval_s=(const float*)d_in[24];
  const float* Wval_v=(const float*)d_in[25];
  const float* bval = (const float*)d_in[26];
  const float* Wo_s = (const float*)d_in[27];
  const float* Wo_v = (const float*)d_in[28];
  const float* bo   = (const float*)d_in[29];
  const float* ln_gs= (const float*)d_in[30];
  const float* ln_bs= (const float*)d_in[31];
  const float* ln_gv= (const float*)d_in[32];
  const float* lna_g= (const float*)d_in[33];
  const float* lna_b= (const float*)d_in[34];

  float* out = (float*)d_out;
  unsigned char* ws = (unsigned char*)d_ws;
  size_t off = 0;
  float* ln_s  = (float*)(ws + off); off += (size_t)NQ * DD * 4;
  float* ln_v  = (float*)(ws + off); off += (size_t)NQ * DD * 3 * 4;
  float* qproj = (float*)(ws + off); off += (size_t)NQ * 512 * 4;
  float* kproj = (float*)(ws + off); off += (size_t)NQ * 512 * 4;
  _Float16* wdtp_h = (_Float16*)(ws + off); off += (size_t)DD * 512 * 2;
  // total workspace use: ~12.7 MB

  cvt_wdtp<<<(DD * 512 + 255) / 256, 256, 0, stream>>>(Wdtp, wdtp_h);
  ln_proj<<<NQ, DD, 0, stream>>>(query_s, query_v, key_s, key_v,
                                 Wq_s, Wq_v, bq, Wk_s, Wk_v, bk,
                                 ln_gs, ln_bs, ln_gv,
                                 ln_s, ln_v, qproj, kproj);
  const size_t smem = 16384 + 8192 + (size_t)KN * 512 * 4 + 512 * 4 + 512 * 4; // 94208 B
  fused_attn<<<NQ, 256, smem, stream>>>(qproj, kproj, ln_s, ln_v,
                                        eir_s, eir_v, edge_emb, maskp, knn,
                                        wdtp_h, bdtp, Wqk_s, Wqk_v, bqk,
                                        Wagv_s, bagv, Wagv_v, Wattn, lna_g, lna_b,
                                        Wval_s, Wval_v, bval, Wo_s, Wo_v, bo, out);
}
